// GlobalModel_64965675319618
// MI455X (gfx1250) — matrix operand order repaired and checked by reading, not compiled, not run on hardware
//
#include <hip/hip_runtime.h>
#include <hip/hip_bf16.h>

// ---------------- problem constants ----------------
#define B_    128
#define CN_   10
#define T_    24
#define E_    100
#define EM_   32
#define H_    256
#define G4_   1024            // 4*H
#define NROW_ (B_*CN_)        // 1280 LSTM rows
#define NNODE_ (B_*T_*CN_)    // 30720 nodes
#define NEDGE_ (T_*B_*E_)     // 307200 edges
#define GH_   256             // GNN_H

typedef __attribute__((ext_vector_type(16))) _Float16 v16h;
typedef __attribute__((ext_vector_type(8)))  _Float16 v8h;
typedef __attribute__((ext_vector_type(8)))  float    v8f;

#define DI __device__ __forceinline__

DI v8f wmma_f16(v16h a, v16h b, v8f c) {
  return __builtin_amdgcn_wmma_f32_16x16x32_f16(
      /*neg_a=*/false, a, /*neg_b=*/false, b,
      /*c_mod=*/(short)0, c, /*reuse_a=*/false, /*reuse_b=*/false);
}

// A fragment: 16x32 f16 tile, row-major source; caller supplies the row ptr
// for this lane's M row (M = lane&15).  lane<16 holds K {k0..+7, k0+16..+23},
// lane>=16 holds K {k0+8..+15, k0+24..+31}  (ISA 7.12.2).
DI v16h load_frag_a(const _Float16* __restrict__ rowp, int k0, int lane) {
  int koff = k0 + ((lane & 16) ? 8 : 0);
  union { v16h v; v8h h[2]; } u;
  u.h[0] = *(const v8h*)(rowp + koff);
  u.h[1] = *(const v8h*)(rowp + koff + 16);
  return u.v;
}

// B fragment: 32x16 f16 tile from row-major weight W[outN, K] (ldb = K).
// lane column N = nbase + (lane&15); lane<16 holds K {k0..+7, k0+16..+23},
// lane>=16 holds K {k0+8..+15, k0+24..+31} (two 16B loads).
DI v16h load_frag_b(const _Float16* __restrict__ W, int ldb, int nbase,
                    int k0, int lane) {
  int n  = nbase + (lane & 15);
  int kk = k0 + ((lane & 16) ? 8 : 0);
  const _Float16* p = W + (size_t)n * ldb + kk;
  union { v16h v; v8h h[2]; } u;
  u.h[0] = *(const v8h*)(p);
  u.h[1] = *(const v8h*)(p + 16);
  return u.v;
}

// ---------------- prep kernels ----------------
__global__ void cvt_f16_kernel(_Float16* __restrict__ dst,
                               const float* __restrict__ src, int n) {
  int i = blockIdx.x * blockDim.x + threadIdx.x;
  if (i < n) dst[i] = (_Float16)src[i];
}

__global__ void fill_f32_kernel(float* __restrict__ p, int n) {
  int i = blockIdx.x * blockDim.x + threadIdx.x;
  if (i < n) p[i] = 0.0f;
}

__global__ void fill_f16_kernel(_Float16* __restrict__ p, int n) {
  int i = blockIdx.x * blockDim.x + threadIdx.x;
  if (i < n) p[i] = (_Float16)0.0f;
}

__global__ void bias_comb_kernel(const float* __restrict__ bih,
                                 const float* __restrict__ bhh,
                                 float* __restrict__ bc) {
  int i = blockIdx.x * blockDim.x + threadIdx.x;
  if (i < G4_) bc[i] = bih[i] + bhh[i];
}

// split mlp1_W [256,514] into two f16 [256,256] blocks (x-row, x-col parts)
__global__ void split_mlp1_kernel(const float* __restrict__ W,
                                  _Float16* __restrict__ Wa,
                                  _Float16* __restrict__ Wb) {
  int i = blockIdx.x * blockDim.x + threadIdx.x;
  if (i >= 256 * 256) return;
  int o = i >> 8, k = i & 255;
  Wa[i] = (_Float16)W[o * 514 + k];
  Wb[i] = (_Float16)W[o * 514 + 256 + k];
}

// a16[n,t,k] = relu(aqi[n,t]*aqi_W[k] + aqi_b[k])   (f16)
__global__ void embed_kernel(const float* __restrict__ aqi,
                             const float* __restrict__ aW,
                             const float* __restrict__ ab,
                             _Float16* __restrict__ a16) {
  int i = blockIdx.x * blockDim.x + threadIdx.x;
  if (i >= NROW_ * T_ * EM_) return;
  int k = i & 31;
  int nt = i >> 5;                    // n*T + t, matches [B,CN,T] flat
  float v = aqi[nt] * aW[k] + ab[k];
  a16[i] = (_Float16)(v > 0.f ? v : 0.f);
}

// ---------------- LSTM ----------------
// gates[1280,1024] = a_t[1280,32] @ Wih^T + h[1280,256] @ Whh^T + (bih+bhh)
// grid (80, 8), block 256 (8 waves); wave -> 16x16 tile; K = 32 + 256.
__global__ void lstm_gemm_kernel(const _Float16* __restrict__ a16,
                                 const _Float16* __restrict__ h16,
                                 const _Float16* __restrict__ Wih,
                                 const _Float16* __restrict__ Whh,
                                 const float* __restrict__ bc,
                                 float* __restrict__ gates, int t) {
  int lane = threadIdx.x & 31;
  int wave = threadIdx.x >> 5;
  int mbase = blockIdx.x * 16;
  int nbase = (blockIdx.y * 8 + wave) * 16;
  int m = mbase + (lane & 15);

  v8f acc = {};
  { // input contribution, K = 32 exactly -> one WMMA
    const _Float16* rowp = a16 + ((size_t)m * T_ + t) * EM_;
    v16h a = load_frag_a(rowp, 0, lane);
    v16h b = load_frag_b(Wih, EM_, nbase, 0, lane);
    acc = wmma_f16(a, b, acc);
  }
  const _Float16* rowp = h16 + (size_t)m * H_;
#pragma unroll
  for (int kk = 0; kk < H_; kk += 32) {
    v16h a = load_frag_a(rowp, kk, lane);
    v16h b = load_frag_b(Whh, H_, nbase, kk, lane);
    acc = wmma_f16(a, b, acc);
  }
  int n = nbase + (lane & 15);
  float bs = bc[n];
  int mo = (lane & 16) ? 8 : 0;
#pragma unroll
  for (int r = 0; r < 8; ++r)
    gates[(size_t)(mbase + mo + r) * G4_ + n] = acc[r] + bs;
}

// pointwise gate math; writes f16 h-state (next GEMM) and x-layout node buf
__global__ void lstm_update_kernel(const float* __restrict__ gates,
                                   float* __restrict__ cst,
                                   _Float16* __restrict__ h16,
                                   _Float16* __restrict__ xall, int t) {
  int idx = blockIdx.x * blockDim.x + threadIdx.x;
  if (idx >= NROW_ * H_) return;
  int n = idx >> 8;
  int hc = idx & 255;
  const float* g = gates + (size_t)n * G4_;
  float gi = g[hc], gf = g[256 + hc], gg = g[512 + hc], go = g[768 + hc];
  float i = 1.f / (1.f + __expf(-gi));
  float f = 1.f / (1.f + __expf(-gf));
  float o = 1.f / (1.f + __expf(-go));
  float c = f * cst[idx] + i * tanhf(gg);
  cst[idx] = c;
  float h = o * tanhf(c);
  h16[idx] = (_Float16)h;
  int b = n / CN_;
  int ci = n - b * CN_;
  size_t v = (size_t)b * (T_ * CN_) + (size_t)t * CN_ + ci;  // b-major x layout
  xall[v * H_ + hc] = (_Float16)h;
}

// ---------------- GNN ----------------
// msg = relu([x[row] x[col] ea] @ mlp1_W^T + b); scatter-add into agg by col.
// grid (NEDGE_/16, 2), block 256; wave -> 16 edges x 16 outputs, 16 WMMAs.
__global__ void mlp1_kernel(const _Float16* __restrict__ xall,
                            const _Float16* __restrict__ W1a,
                            const _Float16* __restrict__ W1b,
                            const int* __restrict__ conn,
                            const float* __restrict__ cityw,
                            const float* __restrict__ mlp1W,
                            const float* __restrict__ mlp1b,
                            float* __restrict__ agg) {
  int lane = threadIdx.x & 31;
  int wave = threadIdx.x >> 5;
  int ebase = blockIdx.x * 16;
  int nbase = (blockIdx.y * 8 + wave) * 16;

  // this lane's edge (for A fragments): indices used as raw flat offsets,
  // faithfully reproducing the reference's (t*B+b)*CN+c -> b-major quirk.
  int em = ebase + (lane & 15);
  int g = em / E_;
  int e = em - g * E_;
  int bb = g & (B_ - 1);
  int r  = conn[(bb * E_ + e) * 2 + 0] + g * CN_;
  int cc = conn[(bb * E_ + e) * 2 + 1] + g * CN_;
  const _Float16* rowp = xall + (size_t)r * H_;
  const _Float16* colp = xall + (size_t)cc * H_;

  v8f acc = {};
#pragma unroll
  for (int kk = 0; kk < H_; kk += 32) {
    v16h a = load_frag_a(rowp, kk, lane);
    v16h b = load_frag_b(W1a, H_, nbase, kk, lane);
    acc = wmma_f16(a, b, acc);
  }
#pragma unroll
  for (int kk = 0; kk < H_; kk += 32) {
    v16h a = load_frag_a(colp, kk, lane);
    v16h b = load_frag_b(W1b, H_, nbase, kk, lane);
    acc = wmma_f16(a, b, acc);
  }

  int n = nbase + (lane & 15);
  float w512 = mlp1W[n * 514 + 512];
  float w513 = mlp1W[n * 514 + 513];
  float bias = mlp1b[n];
  int mo = (lane & 16) ? 8 : 0;
#pragma unroll
  for (int rr = 0; rr < 8; ++rr) {
    int emr = ebase + mo + rr;
    int g2 = emr / E_;
    int e2 = emr - g2 * E_;
    int b2 = g2 & (B_ - 1);
    int colv = conn[(b2 * E_ + e2) * 2 + 1] + g2 * CN_;
    float ea0 = cityw[(size_t)emr * 2 + 0];
    float ea1 = cityw[(size_t)emr * 2 + 1];
    float v = acc[rr] + ea0 * w512 + ea1 * w513 + bias;
    v = v > 0.f ? v : 0.f;
    atomicAdd(&agg[(size_t)colv * GH_ + n], v);
  }
}

__global__ void cnt_kernel(const int* __restrict__ conn, float* __restrict__ cnt) {
  int j = blockIdx.x * blockDim.x + threadIdx.x;
  if (j >= NEDGE_) return;
  int g = j / E_;
  int e = j - g * E_;
  int bb = g & (B_ - 1);
  int colv = conn[(bb * E_ + e) * 2 + 1] + g * CN_;
  atomicAdd(&cnt[colv], 1.0f);
}

__global__ void aggdiv_kernel(const float* __restrict__ agg,
                              const float* __restrict__ cnt,
                              _Float16* __restrict__ agg16) {
  int i = blockIdx.x * blockDim.x + threadIdx.x;
  if (i >= NNODE_ * GH_) return;
  float c = cnt[i >> 8];
  c = c < 1.f ? 1.f : c;
  agg16[i] = (_Float16)(agg[i] / c);
}

// out = relu([x agg] @ mlp2_W^T + b2) ; grid (NNODE_/16, 2), block 256.
__global__ void mlp2_kernel(const _Float16* __restrict__ xall,
                            const _Float16* __restrict__ agg16,
                            const _Float16* __restrict__ W2,
                            const float* __restrict__ b2,
                            float* __restrict__ out) {
  int lane = threadIdx.x & 31;
  int wave = threadIdx.x >> 5;
  int mbase = blockIdx.x * 16;
  int nbase = (blockIdx.y * 8 + wave) * 16;
  int m = mbase + (lane & 15);
  const _Float16* xr = xall + (size_t)m * H_;
  const _Float16* ar = agg16 + (size_t)m * GH_;

  v8f acc = {};
#pragma unroll
  for (int kk = 0; kk < H_; kk += 32) {
    v16h a = load_frag_a(xr, kk, lane);
    v16h b = load_frag_b(W2, 512, nbase, kk, lane);
    acc = wmma_f16(a, b, acc);
  }
#pragma unroll
  for (int kk = 0; kk < GH_; kk += 32) {
    v16h a = load_frag_a(ar, kk, lane);
    v16h b = load_frag_b(W2, 512, nbase, 256 + kk, lane);
    acc = wmma_f16(a, b, acc);
  }
  int n = nbase + (lane & 15);
  float bias = b2[n];
  int mo = (lane & 16) ? 8 : 0;
#pragma unroll
  for (int r = 0; r < 8; ++r) {
    float v = acc[r] + bias;
    v = v > 0.f ? v : 0.f;
    out[(size_t)(mbase + mo + r) * GH_ + n] = v;
  }
}

// ---------------- launcher ----------------
extern "C" void kernel_launch(void* const* d_in, const int* in_sizes, int n_in,
                              void* d_out, int out_size, void* d_ws, size_t ws_size,
                              hipStream_t stream) {
  const float* city_aqi = (const float*)d_in[0];
  const int*   conn     = (const int*)d_in[1];
  const float* city_w   = (const float*)d_in[2];
  const float* aqi_W    = (const float*)d_in[4];
  const float* aqi_b    = (const float*)d_in[5];
  const float* Wih      = (const float*)d_in[6];
  const float* Whh      = (const float*)d_in[7];
  const float* bih      = (const float*)d_in[8];
  const float* bhh      = (const float*)d_in[9];
  const float* mlp1W    = (const float*)d_in[10];
  const float* mlp1b    = (const float*)d_in[11];
  const float* mlp2W    = (const float*)d_in[12];
  const float* mlp2b    = (const float*)d_in[13];
  float* out = (float*)d_out;

  // workspace carve-up (256B aligned regions)
  char* base = (char*)d_ws;
  size_t off = 0;
  auto take = [&](size_t bytes) -> char* {
    char* p = base + off;
    off = (off + bytes + 255) & ~(size_t)255;
    return p;
  };
  _Float16* Wih16 = (_Float16*)take((size_t)G4_ * EM_ * 2);
  _Float16* Whh16 = (_Float16*)take((size_t)G4_ * H_ * 2);
  float*    bc    = (float*)take((size_t)G4_ * 4);
  _Float16* W1a   = (_Float16*)take((size_t)256 * 256 * 2);
  _Float16* W1b   = (_Float16*)take((size_t)256 * 256 * 2);
  _Float16* W2    = (_Float16*)take((size_t)256 * 512 * 2);
  _Float16* a16   = (_Float16*)take((size_t)NROW_ * T_ * EM_ * 2);
  _Float16* h16   = (_Float16*)take((size_t)NROW_ * H_ * 2);
  float*    cst   = (float*)take((size_t)NROW_ * H_ * 4);
  float*    gates = (float*)take((size_t)NROW_ * G4_ * 4);
  _Float16* xall  = (_Float16*)take((size_t)NNODE_ * H_ * 2);
  float*    agg   = (float*)take((size_t)NNODE_ * GH_ * 4);
  float*    cnt   = (float*)take((size_t)NNODE_ * 4);
  _Float16* agg16 = (_Float16*)take((size_t)NNODE_ * GH_ * 2);

  const int TPB = 256;
  auto nb = [](long long n) { return (unsigned)((n + 255) / 256); };

  // prep
  cvt_f16_kernel<<<nb(G4_ * EM_), TPB, 0, stream>>>(Wih16, Wih, G4_ * EM_);
  cvt_f16_kernel<<<nb(G4_ * H_), TPB, 0, stream>>>(Whh16, Whh, G4_ * H_);
  cvt_f16_kernel<<<nb(256 * 512), TPB, 0, stream>>>(W2, mlp2W, 256 * 512);
  split_mlp1_kernel<<<nb(256 * 256), TPB, 0, stream>>>(mlp1W, W1a, W1b);
  bias_comb_kernel<<<nb(G4_), TPB, 0, stream>>>(bih, bhh, bc);
  embed_kernel<<<nb((long long)NROW_ * T_ * EM_), TPB, 0, stream>>>(
      city_aqi, aqi_W, aqi_b, a16);
  fill_f16_kernel<<<nb(NROW_ * H_), TPB, 0, stream>>>(h16, NROW_ * H_);
  fill_f32_kernel<<<nb(NROW_ * H_), TPB, 0, stream>>>(cst, NROW_ * H_);
  fill_f32_kernel<<<nb((long long)NNODE_ * GH_), TPB, 0, stream>>>(agg, NNODE_ * GH_);
  fill_f32_kernel<<<nb(NNODE_), TPB, 0, stream>>>(cnt, NNODE_);

  // LSTM: 24 sequential steps (gemm + pointwise)
  for (int t = 0; t < T_; ++t) {
    lstm_gemm_kernel<<<dim3(NROW_ / 16, G4_ / 128), TPB, 0, stream>>>(
        a16, h16, Wih16, Whh16, bc, gates, t);
    lstm_update_kernel<<<nb(NROW_ * H_), TPB, 0, stream>>>(gates, cst, h16, xall, t);
  }

  // GNN
  mlp1_kernel<<<dim3(NEDGE_ / 16, 2), TPB, 0, stream>>>(
      xall, W1a, W1b, conn, city_w, mlp1W, mlp1b, agg);
  cnt_kernel<<<nb(NEDGE_), TPB, 0, stream>>>(conn, cnt);
  aggdiv_kernel<<<nb((long long)NNODE_ * GH_), TPB, 0, stream>>>(agg, cnt, agg16);
  mlp2_kernel<<<dim3(NNODE_ / 16, 2), TPB, 0, stream>>>(xall, agg16, W2, mlp2b, out);
}
